// rago_9689446220018
// MI455X (gfx1250) — matrix operand order repaired and checked by reading, not compiled, not run on hardware
//
#include <hip/hip_runtime.h>
#include <hip/hip_bf16.h>
#include <math.h>

// ---------------------------------------------------------------------------
// CDNA5 / gfx1250: template-specialized fused gather-GEMM (WMMA
// f32_16x16x32_f16) + RAGO forward pipeline.
//  - A-fragments stream straight from global as b128 runs (gather/concat/r*h
//    fused into the A path; index loads hoisted out of the K-loop)
//  - B-fragments in LDS, fragment-major (2x ds_load_b128 each)
//  - uniform full-tile fast path: no per-row guards
//  - epilogue: wave-private LDS transpose -> coalesced global_store_b128
// ---------------------------------------------------------------------------

typedef __attribute__((ext_vector_type(16))) _Float16 v16h;
typedef __attribute__((ext_vector_type(8)))  _Float16 v8h;
typedef __attribute__((ext_vector_type(8)))  float    v8f;
typedef __attribute__((ext_vector_type(4)))  float    f32x4;
typedef float f32x4u __attribute__((ext_vector_type(4), aligned(4)));

#define HD      48
#define BM      128       // rows per block = 8 waves x 16
#define BSTRIDE 40        // halves per B fragment slot (80B, 16B-aligned)
#define OSTR    52        // dwords per row in output-transpose LDS (16B-aligned)

template <bool B> struct BoolC { static constexpr bool value = B; };

__device__ __forceinline__ float actf(float v, int ACT) {
  switch (ACT) {
    case 1: return v > 0.f ? v : 0.f;
    case 2: return tanhf(v);
    case 3: return 1.f / (1.f + expf(-v));
  }
  return v;
}

// 8 consecutive virtual-A elements starting at ka (compile-time part widths).
template <int W0, int W1, int W2, bool MUL>
__device__ __forceinline__ void a_run8(int ka, const float* __restrict__ b0,
                                       const float* __restrict__ mb,
                                       const float* __restrict__ b1,
                                       const float* __restrict__ b2,
                                       float* f)
{
  constexpr int K = W0 + W1 + W2;
  if (ka >= K) {
#pragma unroll
    for (int j = 0; j < 8; ++j) f[j] = 0.f;
    return;
  }
  if (ka + 8 <= W0) {
    const f32x4u* p = (const f32x4u*)(b0 + ka);
    f32x4u lo = p[0], hi = p[1];
    if (MUL) {
      const f32x4u* q = (const f32x4u*)(mb + ka);
      f32x4u a = q[0], b = q[1];
      lo *= a; hi *= b;
    }
#pragma unroll
    for (int j = 0; j < 4; ++j) { f[j] = lo[j]; f[4 + j] = hi[j]; }
    return;
  }
  if (W1 > 0 && ka >= W0 && ka - W0 + 8 <= W1) {
    const f32x4u* p = (const f32x4u*)(b1 + (ka - W0));
    f32x4u lo = p[0], hi = p[1];
#pragma unroll
    for (int j = 0; j < 4; ++j) { f[j] = lo[j]; f[4 + j] = hi[j]; }
    return;
  }
  if (W2 > 0 && ka >= W0 + W1 && ka - W0 - W1 + 8 <= W2) {
    const f32x4u* p = (const f32x4u*)(b2 + (ka - W0 - W1));
    f32x4u lo = p[0], hi = p[1];
#pragma unroll
    for (int j = 0; j < 4; ++j) { f[j] = lo[j]; f[4 + j] = hi[j]; }
    return;
  }
  // straddling run (only 9-wide parts): scalar, statically pruned chain
#pragma unroll
  for (int j = 0; j < 8; ++j) {
    int k = ka + j;
    float v = 0.f;
    if (k < K) {
      if (k < W0) {
        v = b0[k];
        if (MUL) v *= mb[k];
      } else if (W1 > 0 && k < W0 + W1) {
        v = b1[k - W0];
      } else if (W2 > 0) {
        v = b2[k - W0 - W1];
      }
    }
    f[j] = v;
  }
}

// Y[M,NN] = ACT( concat(A0[i0[r]](*m0), A1[i1[r]], A2[r]) @ W + bias )
template <int W0, int W1, int W2, bool HI0, bool HI1, bool MUL, int NN, int ACT>
__launch_bounds__(256)
__global__ void k_gemm(const float* __restrict__ A0, const int* __restrict__ i0,
                       const float* __restrict__ m0,
                       const float* __restrict__ A1, const int* __restrict__ i1,
                       const float* __restrict__ A2,
                       const float* __restrict__ W, const float* __restrict__ bias,
                       float* __restrict__ Y, int M)
{
  constexpr int K     = W0 + W1 + W2;
  constexpr int KPAD  = (K + 31) & ~31;
  constexpr int NKS   = KPAD / 32;
  constexpr int TILES = (NN + 15) / 16;

  __shared__ _Float16 sB[NKS * TILES * 32 * BSTRIDE];
  __shared__ __attribute__((aligned(16))) float sOut[(NN == 48) ? (8 * 16 * OSTR) : 4];

  const int tid  = threadIdx.x;
  const int wv   = tid >> 5;
  const int lane = tid & 31;
  const int rowBase = blockIdx.x * BM;

  // Stage B fragments (f32 -> f16), fragment-major: frag[j] = W[kb + 8h + (j&7) + 16*(j>>3)][n].
  for (int i = tid; i < NKS * TILES * 32 * 16; i += 256) {
    int j  = i & 15;
    int l  = (i >> 4) & 31;
    int rs = i >> 9;
    int tt = rs % TILES, ks = rs / TILES;
    int k  = (ks << 5) + ((l >> 4) << 3) + (j & 7) + ((j >> 3) << 4);
    int n  = (tt << 4) + (l & 15);
    float v = (k < K && n < NN) ? W[(size_t)k * NN + n] : 0.f;
    sB[(((ks * TILES + tt) << 5) + l) * BSTRIDE + j] = (_Float16)v;
  }
  __syncthreads();

  const int row = rowBase + (wv << 4) + (lane & 15);  // A row for this lane
  const int kh  = (lane >> 4) << 3;                   // A K-half select
  const int nl  = lane & 15;                          // C/D column in tile

  auto run = [&](auto FULLC) {
    constexpr bool FULL = decltype(FULLC)::value;
    const bool rowOK = FULL ? true : (row < M);
    const int rsafe  = FULL ? row : (rowOK ? row : 0);

    // Hoist gather indices / part base pointers out of the K-loop.
    const int r0 = HI0 ? i0[rsafe] : rsafe;
    const int r1 = HI1 ? i1[rsafe] : rsafe;
    const float* b0 = A0 + (size_t)r0 * W0;
    const float* mb = MUL ? (m0 + (size_t)r0 * W0) : nullptr;
    const float* b1 = (W1 > 0) ? (A1 + (size_t)r1 * W1) : nullptr;
    const float* b2 = (W2 > 0) ? (A2 + (size_t)rsafe * W2) : nullptr;

    if (rowOK) __builtin_prefetch(b0, 0, 1);

    v8f acc[TILES];
#pragma unroll
    for (int t = 0; t < TILES; ++t) acc[t] = (v8f){};

#pragma unroll
    for (int ks = 0; ks < NKS; ++ks) {
      const int k0 = ks << 5;
      float f[16];
      if (FULL || rowOK) {
        a_run8<W0, W1, W2, MUL>(k0 + kh,      b0, mb, b1, b2, f);
        a_run8<W0, W1, W2, MUL>(k0 + 16 + kh, b0, mb, b1, b2, f + 8);
      } else {
#pragma unroll
        for (int j = 0; j < 16; ++j) f[j] = 0.f;
      }
      v16h a;
#pragma unroll
      for (int j = 0; j < 16; ++j) a[j] = (_Float16)f[j];

#pragma unroll
      for (int t = 0; t < TILES; ++t) {
        const v8h* bp = (const v8h*)(sB + (((ks * TILES + t) << 5) + lane) * BSTRIDE);
        v8h lo = bp[0], hi = bp[1];
        v16h b = __builtin_shufflevector(lo, hi, 0,1,2,3,4,5,6,7,8,9,10,11,12,13,14,15);
        acc[t] = __builtin_amdgcn_wmma_f32_16x16x32_f16(false, a, false, b, (short)0,
                                                        acc[t], false, false);
      }
    }

    // C/D layout: vgpr r -> row r + 8*(lane>=16); col = (lane&15) + 16*t.
    float bv[TILES];
#pragma unroll
    for (int t = 0; t < TILES; ++t) {
      int c = nl + 16 * t;
      bv[t] = (c < NN) ? bias[c] : 0.f;
    }

    if constexpr (FULL && NN == 48) {
      // Wave-private LDS transpose -> coalesced b128 row stores.
      float* so = sOut + wv * 16 * OSTR;
      const int rhi = (lane >> 4) << 3;
#pragma unroll
      for (int r = 0; r < 8; ++r) {
        int rr = r + rhi;
#pragma unroll
        for (int t = 0; t < TILES; ++t)
          so[rr * OSTR + nl + 16 * t] = actf(acc[t][r] + bv[t], ACT);
      }
      // LDS ops are in-order within a wave; read back rows and store 96B each.
      const int rr2 = lane & 15;
      const int cb  = (lane >> 4) * 24;
      const float* sp = so + rr2 * OSTR + cb;
      float* gp = Y + (size_t)(rowBase + (wv << 4) + rr2) * NN + cb;
#pragma unroll
      for (int i = 0; i < 6; ++i) {
        f32x4 v = *(const f32x4*)(sp + 4 * i);
        *(f32x4*)(gp + 4 * i) = v;
      }
    } else {
      const int mrow = rowBase + (wv << 4) + ((lane >> 4) << 3);
#pragma unroll
      for (int r = 0; r < 8; ++r) {
        int orow = mrow + r;
        if (!FULL && orow >= M) continue;
        float* yrow = Y + (size_t)orow * NN;
#pragma unroll
        for (int t = 0; t < TILES; ++t) {
          int c = nl + 16 * t;
          if (c < NN) yrow[c] = actf(acc[t][r] + bv[t], ACT);
        }
      }
    }
  };

  if (rowBase + BM <= M) run(BoolC<true>{});
  else                   run(BoolC<false>{});
}

// ---------------------------------------------------------------------------
// Elementwise / scatter helpers
// ---------------------------------------------------------------------------

__global__ void k_zero(float* p, size_t n) {
  size_t i = (size_t)blockIdx.x * blockDim.x + threadIdx.x;
  if (i < n) p[i] = 0.f;
}

__global__ void k_tanh_ip(float* p, size_t n) {
  size_t i = (size_t)blockIdx.x * blockDim.x + threadIdx.x;
  if (i < n) p[i] = tanhf(p[i]);
}

__global__ void k_fill6(float* e6, int E) {
  int i = blockIdx.x * blockDim.x + threadIdx.x;
  if (i >= E * 6) return;
  int c = i % 6;
  e6[i] = (c == 0 || c == 4) ? 1.f : 0.f;
}

__global__ void k_seg_accum(const float* __restrict__ v, const int* __restrict__ idx,
                            float* __restrict__ sum, float* __restrict__ cnt, int E, int d)
{
  size_t i = (size_t)blockIdx.x * blockDim.x + threadIdx.x;
  if (i >= (size_t)E * d) return;
  int e = (int)(i / d), j = (int)(i % d);
  int n = idx[e];
  atomicAdd(&sum[(size_t)n * d + j], v[i]);
  if (j == 0) atomicAdd(&cnt[n], 1.f);
}

__global__ void k_seg_div(const float* __restrict__ sum, const float* __restrict__ cnt,
                          float* __restrict__ out, int n, int d)
{
  size_t i = (size_t)blockIdx.x * blockDim.x + threadIdx.x;
  if (i >= (size_t)n * d) return;
  out[i] = sum[i] / fmaxf(cnt[i / d], 1.f);
}

// h = tanh((1-z)*h + z*q)
__global__ void k_gru_out(float* __restrict__ h, const float* __restrict__ z,
                          const float* __restrict__ q, int M, int hd)
{
  size_t i = (size_t)blockIdx.x * blockDim.x + threadIdx.x;
  if (i >= (size_t)M * hd) return;
  float zz = z[i];
  h[i] = tanhf((1.f - zz) * h[i] + zz * q[i]);
}

__device__ __forceinline__ void ortho6d_dev(const float* v, float* Rm) {
  float a1[3] = {v[0], v[1], v[2]}, a2[3] = {v[3], v[4], v[5]};
  float n1 = fmaxf(sqrtf(a1[0]*a1[0] + a1[1]*a1[1] + a1[2]*a1[2]), 1e-8f);
  float b1[3] = {a1[0]/n1, a1[1]/n1, a1[2]/n1};
  float dp = b1[0]*a2[0] + b1[1]*a2[1] + b1[2]*a2[2];
  float t[3] = {a2[0]-dp*b1[0], a2[1]-dp*b1[1], a2[2]-dp*b1[2]};
  float n2 = fmaxf(sqrtf(t[0]*t[0] + t[1]*t[1] + t[2]*t[2]), 1e-8f);
  float b2[3] = {t[0]/n2, t[1]/n2, t[2]/n2};
  float b3[3] = {b1[1]*b2[2]-b1[2]*b2[1], b1[2]*b2[0]-b1[0]*b2[2], b1[0]*b2[1]-b1[1]*b2[0]};
  Rm[0]=b1[0]; Rm[1]=b2[0]; Rm[2]=b3[0];
  Rm[3]=b1[1]; Rm[4]=b2[1]; Rm[5]=b3[1];
  Rm[6]=b1[2]; Rm[7]=b2[2]; Rm[8]=b3[2];
}

__global__ void k_graph_cost(const float* __restrict__ R, const float* __restrict__ e6,
                             const float* __restrict__ A, const int* __restrict__ src,
                             const int* __restrict__ dst, float* __restrict__ ec,
                             float* __restrict__ ncsum, float* __restrict__ nccnt, int E)
{
  int e = blockIdx.x * blockDim.x + threadIdx.x;
  if (e >= E) return;
  int s = src[e], d = dst[e];
  float Rs[9], Re[9], Ae[9];
#pragma unroll
  for (int i = 0; i < 9; ++i) {
    Rs[i] = R[(size_t)s * 9 + i];
    Re[i] = R[(size_t)d * 9 + i];
    Ae[i] = A[(size_t)e * 9 + i];
  }
#pragma unroll
  for (int i = 0; i < 3; ++i)
#pragma unroll
    for (int k = 0; k < 3; ++k) {
      float v = Ae[0*3+i]*Rs[0*3+k] + Ae[1*3+i]*Rs[1*3+k] + Ae[2*3+i]*Rs[2*3+k];
      atomicAdd(&ncsum[(size_t)d * 9 + i*3 + k], fabsf(v - Re[i*3 + k]));
    }
  atomicAdd(&nccnt[d], 1.f);

  float Rm[9];
  ortho6d_dev(&e6[(size_t)e * 6], Rm);
#pragma unroll
  for (int i = 0; i < 3; ++i)
#pragma unroll
    for (int k = 0; k < 3; ++k) {
      float nr = Rs[i*3+0]*Re[k*3+0] + Rs[i*3+1]*Re[k*3+1] + Rs[i*3+2]*Re[k*3+2];
      int o = i*3 + k;
      ec[(size_t)e * 9 + o] = fabsf(Rm[o] - Ae[o]) + fabsf(Rm[o] - nr);
    }
}

__global__ void k_add_rect(float* __restrict__ e6, const float* __restrict__ d6,
                           float* __restrict__ rect, int n)
{
  size_t i = (size_t)blockIdx.x * blockDim.x + threadIdx.x;
  if (i >= (size_t)n) return;
  float v = e6[i] + d6[i];
  e6[i] = v;
  rect[i] = v;
}

__global__ void k_node_update(float* __restrict__ R, const float* __restrict__ v6,
                              float* __restrict__ est, int Nn)
{
  int n = blockIdx.x * blockDim.x + threadIdx.x;
  if (n >= Nn) return;
  float dR[9];
  ortho6d_dev(&v6[(size_t)n * 6], dR);
  float Ro[9], Rc[9];
#pragma unroll
  for (int i = 0; i < 9; ++i) Ro[i] = R[(size_t)n * 9 + i];
#pragma unroll
  for (int i = 0; i < 3; ++i)
#pragma unroll
    for (int k = 0; k < 3; ++k)
      Rc[i*3+k] = Ro[i*3+0]*dR[0*3+k] + Ro[i*3+1]*dR[1*3+k] + Ro[i*3+2]*dR[2*3+k];
#pragma unroll
  for (int i = 0; i < 9; ++i) R[(size_t)n * 9 + i] = Rc[i];

  float m00=Rc[0], m01=Rc[1], m02=Rc[2];
  float m10=Rc[3], m11=Rc[4], m12=Rc[5];
  float m20=Rc[6], m21=Rc[7], m22=Rc[8];
  float s0 = 1.f + m00 + m11 + m22, s1 = 1.f + m00 - m11 - m22;
  float s2 = 1.f - m00 + m11 - m22, s3 = 1.f - m00 - m11 + m22;
  int im = 0; float best = s0;
  if (s1 > best) { best = s1; im = 1; }
  if (s2 > best) { best = s2; im = 2; }
  if (s3 > best) { best = s3; im = 3; }
  float q0, q1, q2, q3;
  if      (im == 0) { q0 = s0;        q1 = m21 - m12; q2 = m02 - m20; q3 = m10 - m01; }
  else if (im == 1) { q0 = m21 - m12; q1 = s1;        q2 = m01 + m10; q3 = m02 + m20; }
  else if (im == 2) { q0 = m02 - m20; q1 = m01 + m10; q2 = s2;        q3 = m12 + m21; }
  else              { q0 = m10 - m01; q1 = m02 + m20; q2 = m12 + m21; q3 = s3;        }
  float nq = fmaxf(sqrtf(q0*q0 + q1*q1 + q2*q2 + q3*q3), 1e-8f);
  float w = q0/nq, x = q1/nq, y = q2/nq, z = q3/nq;
  float* o = est + (size_t)n * 9;
  o[0] = 1.f - 2.f*(y*y + z*z); o[1] = 2.f*(x*y - w*z);       o[2] = 2.f*(x*z + w*y);
  o[3] = 2.f*(x*y + w*z);       o[4] = 1.f - 2.f*(x*x + z*z); o[5] = 2.f*(y*z - w*x);
  o[6] = 2.f*(x*z - w*y);       o[7] = 2.f*(y*z + w*x);       o[8] = 1.f - 2.f*(x*x + y*y);
}

// ---------------------------------------------------------------------------
// Host orchestration
// ---------------------------------------------------------------------------

static inline unsigned gblk(size_t n) { return (unsigned)((n + 255) / 256); }

extern "C" void kernel_launch(void* const* d_in, const int* in_sizes, int n_in,
                              void* d_out, int out_size, void* d_ws, size_t ws_size,
                              hipStream_t stream)
{
  (void)in_sizes; (void)n_in; (void)out_size; (void)ws_size;
  const int Nn = 20000, E = 600000;
  const float* init_rot  = (const float*)d_in[0];
  const float* edge_attr = (const float*)d_in[1];
  const int*   eidx      = (const int*)d_in[2];
  const int* src = eidx;
  const int* dst = eidx + E;
  const unsigned EB = (E + BM - 1) / BM, NB = (Nn + BM - 1) / BM;
  const float* kNull = nullptr;
  const int*   iNull = nullptr;

  auto P = [&](int i) -> const float* { return (const float*)d_in[i]; };
  // Param leaves flattened in sorted-dict-key (jax pytree) order, (w,b) per linear.
  const int EF = 3, EG = 15, EU = 21, MEC = 33, MF = 45, MNC = 49, MS = 61,
            NF = 65, NG = 77, NU = 83;

  char* wsp = (char*)d_ws;
  auto alloc = [&](size_t elems) -> float* {
    float* p = (float*)wsp;
    wsp += ((elems * 4) + 255) & ~(size_t)255;
    return p;
  };
  float* nodeR  = alloc((size_t)Nn * 9);
  float* edge6  = alloc((size_t)E * 6);
  float* nfeat  = alloc((size_t)Nn * HD);
  float* nstate = alloc((size_t)Nn * HD);
  float* efeat  = alloc((size_t)E * HD);
  float* estate = alloc((size_t)E * HD);
  float* ecbuf  = alloc((size_t)E * 9);
  float* ncbuf  = alloc((size_t)Nn * 9);
  float* segsum = alloc((size_t)Nn * HD);
  float* segcnt = alloc((size_t)Nn);
  float* aggN   = alloc((size_t)Nn * HD);
  float* bufEb  = alloc((size_t)E * HD);
  float* bufEc  = alloc((size_t)E * HD);
  float* bufEd  = alloc((size_t)E * HD);
  float* bufEe  = alloc((size_t)E * HD);
  float* bufNb  = alloc((size_t)Nn * HD);
  float* bufNc  = alloc((size_t)Nn * HD);
  float* bufNd  = alloc((size_t)Nn * HD);
  float* bufNe  = alloc((size_t)Nn * HD);

  float* rect = (float*)d_out;                 // [3, E, 6]
  float* est  = rect + (size_t)3 * E * 6;      // [12, Nn, 9]

  auto zero = [&](float* p, size_t n) {
    k_zero<<<gblk(n), 256, 0, stream>>>(p, n);
  };
  auto segmean = [&](const float* v, int d, float* out) {
    zero(segsum, (size_t)Nn * d); zero(segcnt, (size_t)Nn);
    k_seg_accum<<<gblk((size_t)E * d), 256, 0, stream>>>(v, dst, segsum, segcnt, E, d);
    k_seg_div<<<gblk((size_t)Nn * d), 256, 0, stream>>>(segsum, segcnt, out, Nn, d);
  };
  auto conv9 = [&](int pb, const float* x, const float* e, float* m_out, float* xn_out) {
    k_gemm<9, 9, 9, true, true, false, 48, 1><<<EB, 256, 0, stream>>>(
        x, src, kNull, x, dst, e, P(pb), P(pb + 1), m_out, E);
    segmean(m_out, HD, aggN);
    k_gemm<9, 48, 0, false, false, false, 48, 1><<<NB, 256, 0, stream>>>(
        x, iNull, kNull, aggN, iNull, kNull, P(pb + 2), P(pb + 3), xn_out, Nn);
  };
  auto conv48 = [&](int pb, const float* x, const float* e, float* m_out, float* xn_out) {
    k_gemm<48, 48, 48, true, true, false, 48, 1><<<EB, 256, 0, stream>>>(
        x, src, kNull, x, dst, e, P(pb), P(pb + 1), m_out, E);
    segmean(m_out, HD, aggN);
    k_gemm<48, 48, 0, false, false, false, 48, 1><<<NB, 256, 0, stream>>>(
        x, iNull, kNull, aggN, iNull, kNull, P(pb + 2), P(pb + 3), xn_out, Nn);
  };
  auto mlp6_2 = [&](int pb, const float* xa, const float* xb, unsigned MBk, int M,
                    float* h1, float* h2, float* out) {
    k_gemm<48, 48, 0, false, false, false, 48, 1><<<MBk, 256, 0, stream>>>(
        xa, iNull, kNull, xb, iNull, kNull, P(pb + 0), P(pb + 1), h1, M);
    k_gemm<48, 0, 0, false, false, false, 48, 1><<<MBk, 256, 0, stream>>>(
        h1, iNull, kNull, kNull, iNull, kNull, P(pb + 2), P(pb + 3), h2, M);
    k_gemm<48, 0, 0, false, false, false, 48, 1><<<MBk, 256, 0, stream>>>(
        h2, iNull, kNull, kNull, iNull, kNull, P(pb + 4), P(pb + 5), h1, M);
    k_gemm<48, 48, 48, false, false, false, 48, 1><<<MBk, 256, 0, stream>>>(
        h1, iNull, kNull, xa, iNull, xb, P(pb + 6), P(pb + 7), h2, M);
    k_gemm<48, 0, 0, false, false, false, 48, 1><<<MBk, 256, 0, stream>>>(
        h2, iNull, kNull, kNull, iNull, kNull, P(pb + 8), P(pb + 9), h1, M);
    k_gemm<48, 0, 0, false, false, false, 48, 0><<<MBk, 256, 0, stream>>>(
        h1, iNull, kNull, kNull, iNull, kNull, P(pb + 10), P(pb + 11), out, M);
  };
  auto mlp6_1 = [&](int pb, const float* xa, unsigned MBk, int M,
                    float* h1, float* h2, float* out) {
    k_gemm<48, 0, 0, false, false, false, 48, 1><<<MBk, 256, 0, stream>>>(
        xa, iNull, kNull, kNull, iNull, kNull, P(pb + 0), P(pb + 1), h1, M);
    k_gemm<48, 0, 0, false, false, false, 48, 1><<<MBk, 256, 0, stream>>>(
        h1, iNull, kNull, kNull, iNull, kNull, P(pb + 2), P(pb + 3), h2, M);
    k_gemm<48, 0, 0, false, false, false, 48, 1><<<MBk, 256, 0, stream>>>(
        h2, iNull, kNull, kNull, iNull, kNull, P(pb + 4), P(pb + 5), h1, M);
    k_gemm<48, 48, 0, false, false, false, 48, 1><<<MBk, 256, 0, stream>>>(
        h1, iNull, kNull, xa, iNull, kNull, P(pb + 6), P(pb + 7), h2, M);
    k_gemm<48, 0, 0, false, false, false, 48, 1><<<MBk, 256, 0, stream>>>(
        h2, iNull, kNull, kNull, iNull, kNull, P(pb + 8), P(pb + 9), h1, M);
    k_gemm<48, 0, 0, false, false, false, 6, 0><<<MBk, 256, 0, stream>>>(
        h1, iNull, kNull, kNull, iNull, kNull, P(pb + 10), P(pb + 11), out, M);
  };
  auto gru = [&](int pb, float* h, const float* xa, const float* xb, unsigned MBk, int M,
                 float* bz, float* br, float* bq) {
    k_gemm<48, 48, 48, false, false, false, 48, 3><<<MBk, 256, 0, stream>>>(
        h, iNull, kNull, xa, iNull, xb, P(pb + 4), P(pb + 5), bz, M);   // z
    k_gemm<48, 48, 48, false, false, false, 48, 3><<<MBk, 256, 0, stream>>>(
        h, iNull, kNull, xa, iNull, xb, P(pb + 2), P(pb + 3), br, M);   // r
    k_gemm<48, 48, 48, false, false, true, 48, 2><<<MBk, 256, 0, stream>>>(
        h, iNull, br, xa, iNull, xb, P(pb + 0), P(pb + 1), bq, M);      // q
    k_gru_out<<<gblk((size_t)M * HD), 256, 0, stream>>>(h, bz, bq, M, HD);
  };
  auto graphcost = [&]() {
    zero(segsum, (size_t)Nn * 9); zero(segcnt, (size_t)Nn);
    k_graph_cost<<<gblk(E), 256, 0, stream>>>(nodeR, edge6, edge_attr, src, dst,
                                              ecbuf, segsum, segcnt, E);
    k_seg_div<<<gblk((size_t)Nn * 9), 256, 0, stream>>>(segsum, segcnt, ncbuf, Nn, 9);
  };

  // ---- init ----
  hipMemcpyAsync(nodeR, init_rot, (size_t)Nn * 9 * sizeof(float),
                 hipMemcpyDeviceToDevice, stream);
  k_fill6<<<gblk((size_t)E * 6), 256, 0, stream>>>(edge6, E);

  conv9(MF, init_rot, edge_attr, efeat, nfeat);
  conv9(MS, init_rot, edge_attr, estate, nstate);
  k_tanh_ip<<<gblk((size_t)E * HD), 256, 0, stream>>>(estate, (size_t)E * HD);
  k_tanh_ip<<<gblk((size_t)Nn * HD), 256, 0, stream>>>(nstate, (size_t)Nn * HD);

  // ---- main loop: tg=3, te=1, tn=4 ----
  for (int ig = 0; ig < 3; ++ig) {
    // ---- edge iteration ----
    graphcost();
    conv9 (MEC + 0, ncbuf, ecbuf, bufEb, bufNb);            // m1 = bufEb
    conv48(MEC + 4, bufNb, bufEb, bufEc, bufNc);            // m2 = bufEc
    conv48(MEC + 8, bufNc, bufEc, bufEb, bufNb);            // ecf = bufEb
    mlp6_2(EF, bufEb, efeat, EB, E, bufEc, bufEd, bufEd);   // ef = bufEd
    gru(EG, estate, bufEd, efeat, EB, E, bufEb, bufEc, bufEe);
    mlp6_1(EU, estate, EB, E, bufEb, bufEc, bufEd);         // d6 = bufEd[E,6]
    k_add_rect<<<gblk((size_t)E * 6), 256, 0, stream>>>(
        edge6, bufEd, rect + (size_t)ig * E * 6, E * 6);

    // ---- node iterations ----
    for (int ino = 0; ino < 4; ++ino) {
      graphcost();
      conv9 (MNC + 0, ncbuf, ecbuf, bufEb, bufNb);
      conv48(MNC + 4, bufNb, bufEb, bufEc, bufNc);
      conv48(MNC + 8, bufNc, bufEc, bufEb, bufNd);          // ncf = bufNd
      mlp6_2(NF, bufNd, nfeat, NB, Nn, bufNb, bufNc, bufNc);
      gru(NG, nstate, bufNc, nfeat, NB, Nn, bufNb, bufNd, bufNe);
      mlp6_1(NU, nstate, NB, Nn, bufNb, bufNc, bufNd);      // [Nn,6]
      k_node_update<<<gblk(Nn), 256, 0, stream>>>(
          nodeR, bufNd, est + (size_t)(ig * 4 + ino) * Nn * 9, Nn);
    }
  }
}
